// StaticWeightedGATLayer_790273982955
// MI455X (gfx1250) — matrix operand order repaired and checked by reading, not compiled, not run on hardware
//
#include <hip/hip_runtime.h>
#include <hip/hip_bf16.h>

// Problem constants (reference: B=8, N=2048, D=128, H=4, DH=32)
#define Bc  8
#define Nc  2048
#define Dc  128
#define Hc  4
#define DHc 32
#define EPS_LN 1e-5f
#define NEG_BIG (-1e30f)

#define CHUNK   256              // h^T columns staged per TDM transfer
#define NCHUNKS (Nc / CHUNK)     // 8

typedef __bf16 bf16_t;
typedef __attribute__((ext_vector_type(8)))  bf16_t v8bf;
typedef __attribute__((ext_vector_type(16))) bf16_t v16bf;
typedef __attribute__((ext_vector_type(8)))  float  v8f;
typedef __attribute__((ext_vector_type(4)))  unsigned u32x4;
typedef __attribute__((ext_vector_type(4)))  int      i32x4;
typedef __attribute__((ext_vector_type(8)))  int      i32x8;

#if defined(__HIP_DEVICE_COMPILE__) && \
    __has_builtin(__builtin_amdgcn_tensor_load_to_lds) && \
    __has_builtin(__builtin_amdgcn_s_wait_tensorcnt)
#define USE_TDM 1
#else
#define USE_TDM 0
#endif

static __device__ __forceinline__ v8f wmma_bf16(v16bf a, v16bf b, v8f c) {
  // D = A(16x32 bf16) * B(32x16 bf16) + C(16x16 f32)
  return __builtin_amdgcn_wmma_f32_16x16x32_bf16(false, a, false, b,
                                                 (short)0, c, false, false);
}

#if USE_TDM
// Issue one TDM transfer: 2D tile (DHc rows x CHUNK cols, 2B elements) from
// h^T (row stride Nc elements) starting at column jc, into LDS at lds_byte.
// D# bitfields per CDNA5 ISA.
static __device__ __forceinline__ void tdm_issue(const bf16_t* gbase,
                                                 unsigned lds_byte, int jc) {
  unsigned long long ga =
      (unsigned long long)(uintptr_t)gbase + (unsigned long long)jc * 2ull;
  u32x4 g0;
  g0[0] = 1u;                                   // count=1, user mode, no gather
  g0[1] = lds_byte;                             // lds_addr (bytes)
  g0[2] = (unsigned)ga;                         // global_addr[31:0]
  g0[3] = ((unsigned)(ga >> 32) & 0x01FFFFFFu)  // global_addr[56:32]
          | (2u << 30);                         // type = 2 ("image")
  i32x8 g1 = {0, 0, 0, 0, 0, 0, 0, 0};
  g1[0] = (int)(1u << 16);                      // data_size=1 (2 bytes)
  g1[1] = (int)(((unsigned)Nc & 0xFFFFu) << 16);        // tensor_dim0[15:0]
  g1[2] = (int)(((unsigned)Nc >> 16) | ((unsigned)DHc << 16)); // dim0 hi | dim1 lo
  g1[3] = (int)((unsigned)CHUNK << 16);         // tile_dim0
  g1[4] = DHc;                                  // tile_dim1
  g1[5] = Nc;                                   // tensor_dim0_stride[31:0]
  i32x4 g2 = {0, 0, 0, 0};
  i32x4 g3 = {0, 0, 0, 0};
#if __clang_major__ >= 23
  i32x8 g4 = {0, 0, 0, 0, 0, 0, 0, 0};
  __builtin_amdgcn_tensor_load_to_lds(g0, g1, g2, g3, g4, 0);
#else
  __builtin_amdgcn_tensor_load_to_lds(g0, g1, g2, g3, 0);
#endif
}
#endif

// ---------------------------------------------------------------- prep
__global__ void k_cvt_bf16(const float* __restrict__ s, bf16_t* __restrict__ d, int n) {
  int i = blockIdx.x * blockDim.x + threadIdx.x;
  if (i < n) d[i] = (bf16_t)s[i];
}

__global__ void k_loga(const float* __restrict__ adj, float* __restrict__ logA, int n) {
  int i = blockIdx.x * blockDim.x + threadIdx.x;
  if (i < n) {
    float a = adj[i];
    logA[i] = (a > 0.0f) ? __logf(fmaxf(a, 1e-8f)) : NEG_BIG;
  }
}

// ---------------------------------------------------------------- h = x @ W^T  (store transposed (B,H,DH,N) bf16)
__global__ void k_gemm_h(const bf16_t* __restrict__ xbf,
                         const bf16_t* __restrict__ Wbf,
                         bf16_t* __restrict__ hT) {
  const int lane = threadIdx.x & 31;
  const int wave = threadIdx.x >> 5;
  const int it   = blockIdx.x * 16;          // row tile in [0, B*N)
  const int jt   = wave * 16;                // col tile in [0, 128)
  const int m16  = lane & 15;
  const int hi   = lane >> 4;
  const int kb   = hi * 8;
  const int kb16 = hi * 16;

  const int arow = it + m16;
  const int j    = jt + m16;

  v8f acc = {0.f,0.f,0.f,0.f,0.f,0.f,0.f,0.f};
  #pragma unroll
  for (int kt = 0; kt < Dc; kt += 32) {
    const bf16_t* ap = xbf + (size_t)arow * Dc + kt;
    v8bf a0 = *(const v8bf*)(ap + kb16);
    v8bf a1 = *(const v8bf*)(ap + kb16 + 8);
    v16bf A = __builtin_shufflevector(a0, a1, 0,1,2,3,4,5,6,7,8,9,10,11,12,13,14,15);
    v16bf Bm = *(const v16bf*)(Wbf + (size_t)j * Dc + kt + kb16);
    acc = wmma_bf16(A, Bm, acc);
  }

  const int head  = j >> 5;
  const int dfeat = j & 31;
  const int b     = it >> 11;
  #pragma unroll
  for (int v = 0; v < 8; ++v) {
    int grow = it + hi * 8 + v;
    int n    = grow & (Nc - 1);
    hT[(((size_t)b * Hc + head) * DHc + dfeat) * Nc + n] = (bf16_t)acc[v];
  }
}

// ---------------------------------------------------------------- src/dst head projections
__global__ void k_srcdst(const bf16_t* __restrict__ hT,
                         const float* __restrict__ a_src,
                         const float* __restrict__ a_dst,
                         float* __restrict__ src, float* __restrict__ dst) {
  int idx = blockIdx.x * blockDim.x + threadIdx.x;
  if (idx >= Bc * Hc * Nc) return;
  int n  = idx & (Nc - 1);
  int bh = idx >> 11;
  int h  = bh & (Hc - 1);
  const bf16_t* base = hT + (size_t)bh * DHc * Nc + n;
  float s = 0.f, t = 0.f;
  #pragma unroll
  for (int d = 0; d < DHc; ++d) {
    float hv = (float)base[(size_t)d * Nc];
    s += hv * a_src[h * DHc + d];
    t += hv * a_dst[h * DHc + d];
  }
  src[idx] = s;
  dst[idx] = t;
}

// ---------------------------------------------------------------- per-row softmax stats (max, sum)
__global__ void k_stats(const float* __restrict__ src, const float* __restrict__ dst,
                        const float* __restrict__ logA,
                        float* __restrict__ rmax, float* __restrict__ rsum) {
  const int lane  = threadIdx.x & 31;
  const int wave  = threadIdx.x >> 5;
  const int rowid = blockIdx.x * 8 + wave;
  const int bh = rowid >> 11;
  const int i  = rowid & (Nc - 1);
  const float sv = src[rowid];
  const float* dv = dst + (size_t)bh * Nc;
  const float* lr = logA + (size_t)i * Nc;

  float m = NEG_BIG;
  for (int j = lane; j < Nc; j += 32) {
    float t = sv + dv[j];
    t = (t > 0.f) ? t : 0.2f * t;
    m = fmaxf(m, t + lr[j]);
  }
  #pragma unroll
  for (int off = 16; off >= 1; off >>= 1) m = fmaxf(m, __shfl_xor(m, off, 32));

  float s = 0.f;
  for (int j = lane; j < Nc; j += 32) {
    float t = sv + dv[j];
    t = (t > 0.f) ? t : 0.2f * t;
    s += __expf((t + lr[j]) - m);
  }
  #pragma unroll
  for (int off = 16; off >= 1; off >>= 1) s += __shfl_xor(s, off, 32);

  if (lane == 0) { rmax[rowid] = m; rsum[rowid] = s; }
}

// ---------------------------------------------------------------- out = softmax(e) @ h
// All 8 waves of a block share (b,h); h^T tiles are staged into LDS in
// double-buffered 32xCHUNK chunks via the Tensor Data Mover (TDM) when
// available, with s_wait_tensorcnt pipelining; WMMA B operands read from LDS.
__global__ void k_attn_pv(const float* __restrict__ src, const float* __restrict__ dst,
                          const float* __restrict__ logA,
                          const float* __restrict__ rmax, const float* __restrict__ rsum,
                          const bf16_t* __restrict__ hT,
                          bf16_t* __restrict__ aobf) {
  __shared__ __attribute__((aligned(64))) bf16_t hs[2 * DHc * CHUNK]; // 32 KB

  const int lane = threadIdx.x & 31;
  const int wave = threadIdx.x >> 5;
  const int tile = blockIdx.x * 8 + wave;            // B*H*(N/16) tiles
  const int bh   = tile >> 7;                        // shared by all waves in block
  const int it   = (tile & 127) * 16;
  const int b    = bh >> 2;
  const int hd   = bh & (Hc - 1);

  const int m16  = lane & 15;
  const int hi   = lane >> 4;
  const int kb   = hi * 8;
  const int kb16 = hi * 16;

  const int i = it + m16;                            // P row this lane feeds
  const int rowid = bh * Nc + i;
  const float rm   = rmax[rowid];
  const float rinv = 1.0f / rsum[rowid];
  const float sv   = src[rowid];

  const float*  dv  = dst + (size_t)bh * Nc;
  const float*  lr  = logA + (size_t)i * Nc;
  const bf16_t* hTb = hT + (size_t)bh * DHc * Nc;

  v8f acc0 = {0.f,0.f,0.f,0.f,0.f,0.f,0.f,0.f};
  v8f acc1 = {0.f,0.f,0.f,0.f,0.f,0.f,0.f,0.f};

#if USE_TDM
  const unsigned ldsBase = (unsigned)(uintptr_t)(void*)hs;  // addr[31:0] = LDS offset
  if (wave == 0) tdm_issue(hTb, ldsBase, 0);
#endif

  int cbuf = 0;
  for (int c = 0; c < NCHUNKS; ++c) {
    const int jc = c * CHUNK;
#if USE_TDM
    if (wave == 0) {
      if (c + 1 < NCHUNKS) {
        tdm_issue(hTb, ldsBase + (unsigned)((cbuf ^ 1) * DHc * CHUNK * 2),
                  (c + 1) * CHUNK);
        __builtin_amdgcn_s_wait_tensorcnt(1);  // oldest (current buffer) done
      } else {
        __builtin_amdgcn_s_wait_tensorcnt(0);
      }
    }
    __syncthreads();
#else
    { // cooperative synchronous staging fallback (also host-pass body)
      const int tid = threadIdx.x;
      const int d   = tid >> 3;
      const int cb  = (tid & 7) * 32;
      const bf16_t* g = hTb + (size_t)d * Nc + jc + cb;
      bf16_t* l = hs + cbuf * (DHc * CHUNK) + d * CHUNK + cb;
      #pragma unroll
      for (int q = 0; q < 4; ++q)
        *(v8bf*)(l + q * 8) = *(const v8bf*)(g + q * 8);
    }
    __syncthreads();
#endif

    const bf16_t* hb = hs + cbuf * (DHc * CHUNK);
    for (int jl = 0; jl < CHUNK; jl += 32) {
      __builtin_prefetch(lr + jc + jl + 64, 0, 0);   // next logA line
      v16bf A;
      #pragma unroll
      for (int e = 0; e < 16; ++e) {
        int k = kb16 + e;                            // A-operand element->K map
        int j = jc + jl + k;
        float t = sv + dv[j];
        t = (t > 0.f) ? t : 0.2f * t;
        float p = __expf((t + lr[j]) - rm) * rinv;   // masked cols underflow to 0
        A[e] = (bf16_t)p;
      }
      v16bf B0 = *(const v16bf*)(hb + m16 * CHUNK + jl + kb16);
      v16bf B1 = *(const v16bf*)(hb + (16 + m16) * CHUNK + jl + kb16);
      acc0 = wmma_bf16(A, B0, acc0);
      acc1 = wmma_bf16(A, B1, acc1);
    }
    __syncthreads();   // all waves done with cbuf before it is re-staged
    cbuf ^= 1;
  }

  #pragma unroll
  for (int v = 0; v < 8; ++v) {
    int grow = (b << 11) + it + hi * 8 + v;
    size_t o = (size_t)grow * Dc + hd * DHc;
    aobf[o + m16]      = (bf16_t)acc0[v];
    aobf[o + 16 + m16] = (bf16_t)acc1[v];
  }
}

// ---------------------------------------------------------------- y = LN(x + ao @ Wo^T + bo)
__global__ void k_gemm_out_ln(const bf16_t* __restrict__ aobf,
                              const bf16_t* __restrict__ Wobf,
                              const float* __restrict__ x,
                              const float* __restrict__ bo,
                              const float* __restrict__ gamma,
                              const float* __restrict__ beta,
                              float* __restrict__ out) {
  __shared__ float yt[16 * Dc];

  const int lane = threadIdx.x & 31;
  const int wave = threadIdx.x >> 5;
  const int row0 = blockIdx.x * 16;
  const int m16  = lane & 15;
  const int hi   = lane >> 4;
  const int kb   = hi * 8;
  const int kb16 = hi * 16;

  const int arow = row0 + m16;
  const int j    = wave * 16 + m16;

  v8f acc = {0.f,0.f,0.f,0.f,0.f,0.f,0.f,0.f};
  #pragma unroll
  for (int kt = 0; kt < Dc; kt += 32) {
    const bf16_t* ap = aobf + (size_t)arow * Dc + kt;
    v8bf a0 = *(const v8bf*)(ap + kb16);
    v8bf a1 = *(const v8bf*)(ap + kb16 + 8);
    v16bf A = __builtin_shufflevector(a0, a1, 0,1,2,3,4,5,6,7,8,9,10,11,12,13,14,15);
    v16bf Bm = *(const v16bf*)(Wobf + (size_t)j * Dc + kt + kb16);
    acc = wmma_bf16(A, Bm, acc);
  }

  #pragma unroll
  for (int v = 0; v < 8; ++v) {
    int m   = hi * 8 + v;
    int col = wave * 16 + m16;
    float y = acc[v] + x[(size_t)(row0 + m) * Dc + col] + bo[col];
    yt[m * Dc + col] = y;
  }
  __syncthreads();

  const int r = wave * 2 + hi;
  const float* yr = yt + r * Dc;
  float s = 0.f;
  #pragma unroll
  for (int c = 0; c < 8; ++c) s += yr[m16 * 8 + c];
  #pragma unroll
  for (int off = 8; off >= 1; off >>= 1) s += __shfl_xor(s, off, 16);
  const float mu = s * (1.0f / Dc);

  float s2 = 0.f;
  #pragma unroll
  for (int c = 0; c < 8; ++c) { float d = yr[m16 * 8 + c] - mu; s2 += d * d; }
  #pragma unroll
  for (int off = 8; off >= 1; off >>= 1) s2 += __shfl_xor(s2, off, 16);
  const float rstd = rsqrtf(s2 * (1.0f / Dc) + EPS_LN);

  #pragma unroll
  for (int c = 0; c < 8; ++c) {
    int col = m16 * 8 + c;
    out[(size_t)(row0 + r) * Dc + col] =
        (yr[col] - mu) * rstd * gamma[col] + beta[col];
  }
}

// ---------------------------------------------------------------- launcher
extern "C" void kernel_launch(void* const* d_in, const int* in_sizes, int n_in,
                              void* d_out, int out_size, void* d_ws, size_t ws_size,
                              hipStream_t stream) {
  const float* x     = (const float*)d_in[0];   // (B,N,D)
  const float* adj   = (const float*)d_in[1];   // (N,N)
  const float* W     = (const float*)d_in[2];   // (D,D)
  const float* a_src = (const float*)d_in[3];   // (H,DH)
  const float* a_dst = (const float*)d_in[4];   // (H,DH)
  const float* Wo    = (const float*)d_in[5];   // (D,D)
  const float* bo    = (const float*)d_in[6];   // (D,)
  const float* gamma = (const float*)d_in[7];   // (D,)
  const float* beta  = (const float*)d_in[8];   // (D,)
  float* out = (float*)d_out;

  char* ws = (char*)d_ws;
  size_t off = 0;
  auto alloc = [&](size_t bytes) { char* p = ws + off; off = (off + bytes + 255) & ~(size_t)255; return p; };

  bf16_t* xbf  = (bf16_t*)alloc((size_t)Bc * Nc * Dc * sizeof(bf16_t));
  bf16_t* Wbf  = (bf16_t*)alloc((size_t)Dc * Dc * sizeof(bf16_t));
  bf16_t* Wobf = (bf16_t*)alloc((size_t)Dc * Dc * sizeof(bf16_t));
  float*  logA = (float*)alloc((size_t)Nc * Nc * sizeof(float));
  bf16_t* hT   = (bf16_t*)alloc((size_t)Bc * Hc * DHc * Nc * sizeof(bf16_t));
  float*  srcb = (float*)alloc((size_t)Bc * Hc * Nc * sizeof(float));
  float*  dstb = (float*)alloc((size_t)Bc * Hc * Nc * sizeof(float));
  float*  rmax = (float*)alloc((size_t)Bc * Hc * Nc * sizeof(float));
  float*  rsum = (float*)alloc((size_t)Bc * Hc * Nc * sizeof(float));
  bf16_t* aobf = (bf16_t*)alloc((size_t)Bc * Nc * Dc * sizeof(bf16_t));
  (void)ws_size; (void)in_sizes; (void)n_in; (void)out_size;

  const int nx = Bc * Nc * Dc;
  const int nw = Dc * Dc;
  const int na = Nc * Nc;

  k_cvt_bf16<<<(nx + 255) / 256, 256, 0, stream>>>(x,  xbf,  nx);
  k_cvt_bf16<<<(nw + 255) / 256, 256, 0, stream>>>(W,  Wbf,  nw);
  k_cvt_bf16<<<(nw + 255) / 256, 256, 0, stream>>>(Wo, Wobf, nw);
  k_loga   <<<(na + 255) / 256, 256, 0, stream>>>(adj, logA, na);

  k_gemm_h <<<Bc * Nc / 16, 256, 0, stream>>>(xbf, Wbf, hT);

  k_srcdst <<<Bc * Hc * Nc / 256, 256, 0, stream>>>(hT, a_src, a_dst, srcb, dstb);

  k_stats  <<<Bc * Hc * Nc / 8, 256, 0, stream>>>(srcb, dstb, logA, rmax, rsum);

  k_attn_pv<<<Bc * Hc * (Nc / 16) / 8, 256, 0, stream>>>(srcb, dstb, logA, rmax, rsum, hT, aobf);

  k_gemm_out_ln<<<Bc * Nc / 16, 256, 0, stream>>>(aobf, Wobf, x, bo, gamma, beta, out);
}
